// GraphCut_65292092834593
// MI455X (gfx1250) — matrix operand order repaired and checked by reading, not compiled, not run on hardware
//
#include <hip/hip_runtime.h>
#include <hip/hip_bf16.h>

// ---------------------------------------------------------------------------
// GraphCut contrastive loss, MI455X (gfx1250, wave32, WMMA).
//
//   sim = (F F^T)/T ; row max ; neg-label-masked sum of exp(sim - max) ;
//   loss_i = sum / (class_cnt_i - 1) ; mean over iou-kept rows.
//
// Fused pass over the 8192x8192 Gram matrix (online softmax, never
// materialized). GEMM in split-bf16 emulated-fp32 (hi*hi + hi*lo + lo*hi)
// via v_wmma_f32_16x16x32_bf16: 12 WMMAs per 16x16 tile.
//
// Work decomposition: 64 row-blocks x 8 column-chunks = 512 blocks of 8
// waves (4096 waves) so every SIMD32 on the chip carries several waves.
// Each wave owns a 16-row strip over a 1024-column chunk; per-row partial
// (max, sum) pairs are softmax-merged in the final reduction kernel.
//
// Epilogue runs in the native base-2 domain (planes pre-scaled by
// sqrt((1/T)*log2 e) so the WMMA accumulator IS the base-2 logit) and needs
// exactly ONE v_exp_f32 per element, with its argument formed via free
// neg/abs source modifiers -- no fmin/fmax NaN-canonicalization ops.
// ---------------------------------------------------------------------------

#define M_ROWS    8192
#define K_DIM     128
#define N_CLASSES 20
#define IOU_THR   0.5f
#define LAMDA_W   1.0f
// sqrt( (1/0.2) * log2(e) ) = sqrt(7.2134752) -- applied to BOTH operands
#define SCALE_PRE 2.6857914f

#define CCHUNKS   8                         // column-dimension split
#define NTILES    (M_ROWS / 16)             // 512 column tiles total
#define TILES_PC  (NTILES / CCHUNKS)        // 64 tiles per chunk

typedef __attribute__((ext_vector_type(16))) __bf16 v16bf;
typedef __attribute__((ext_vector_type(8)))  __bf16 v8bf;
typedef __attribute__((ext_vector_type(8)))  float  v8f;

union V16 { v16bf v; v8bf h[2]; };

__device__ __forceinline__ float fast_exp2(float x) {
    return __builtin_amdgcn_exp2f(x);          // v_exp_f32 (native 2^x)
}

// ---- workspace layout (bytes) ---------------------------------------------
//  [0,               2 MB)  : hi bf16 plane, [M][K]   (pre-scaled)
//  [2 MB,            4 MB)  : lo bf16 plane, [M][K]   (pre-scaled residual)
//  [4 MB,        4 MB+128)  : int cnt[20] (padded)
//  pm : float[M][CCHUNKS]  partial row maxima   (256 KB)
//  ps : float[M][CCHUNKS]  partial masked sums  (256 KB)
#define WS_CNT_OFF  (4u * 1024u * 1024u)
#define WS_PM_OFF   (WS_CNT_OFF + 128u)
#define WS_PS_OFF   (WS_PM_OFF + (unsigned)(M_ROWS * CCHUNKS * sizeof(float)))

// ---------------------------------------------------------------------------
__global__ void k_init(int* __restrict__ cnt) {
    int t = threadIdx.x;
    if (t < N_CLASSES) cnt[t] = 0;
}

__global__ void k_hist(const int* __restrict__ lab, int* __restrict__ cnt) {
    int i = blockIdx.x * blockDim.x + threadIdx.x;
    if (i < M_ROWS) atomicAdd(&cnt[lab[i]], 1);
}

// split (SCALE_PRE * f32) -> hi bf16 + residual lo bf16
__global__ void k_split(const float* __restrict__ f,
                        __bf16* __restrict__ hi, __bf16* __restrict__ lo) {
    int i = blockIdx.x * blockDim.x + threadIdx.x;
    if (i < M_ROWS * K_DIM) {
        float x = f[i] * SCALE_PRE;
        __bf16 h = (__bf16)x;
        float  r = x - (float)h;
        hi[i] = h;
        lo[i] = (__bf16)r;
    }
}

// ---------------------------------------------------------------------------
// Main fused kernel: 256 threads = 8 waves per block. Block (rowBlk, chunk)
// covers rows [rowBlk*128, +128) x column tiles [chunk*64, +64).
// ---------------------------------------------------------------------------
__global__ __launch_bounds__(256)
void k_main(const __bf16* __restrict__ hi, const __bf16* __restrict__ lo,
            const int*    __restrict__ lab,
            float* __restrict__ pm, float* __restrict__ ps)
{
    const int lane  = threadIdx.x & 31;
    const int wave  = threadIdx.x >> 5;
    const int half  = lane >> 4;     // 0: lanes 0-15, 1: lanes 16-31
    const int ln    = lane & 15;
    const int rowBlk = blockIdx.x >> 3;      // 0..63
    const int chunk  = blockIdx.x & 7;       // 0..7
    const int rowBase = (rowBlk * 8 + wave) * 16;

    // ---- A fragments for this wave's 16 rows (hi & lo planes) -------------
    // A fragment as loaded here (contiguous k order): lane holds row (lane&15); 16 consecutive K
    // values starting at half*16 within each K32 chunk (matches B).
    const int arow = rowBase + ln;
    V16 ah[4], al[4];
#pragma unroll
    for (int c = 0; c < 4; ++c) {
        const size_t kbase = (size_t)arow * K_DIM + c * 32 + half * 16;
        ah[c].h[0] = *(const v8bf*)(hi + kbase);
        ah[c].h[1] = *(const v8bf*)(hi + kbase + 8);
        al[c].h[0] = *(const v8bf*)(lo + kbase);
        al[c].h[1] = *(const v8bf*)(lo + kbase + 8);
    }

    // Row labels for the 8 accumulator rows owned by this lane
    int rlab[8];
#pragma unroll
    for (int r = 0; r < 8; ++r) rlab[r] = lab[rowBase + half * 8 + r];

    float mrun[8], srun[8];
#pragma unroll
    for (int r = 0; r < 8; ++r) { mrun[r] = -__builtin_inff(); srun[r] = 0.0f; }

    // ---- sweep this chunk's column tiles ----------------------------------
    const int ct0 = chunk * TILES_PC;
    for (int ct = ct0; ct < ct0 + TILES_PC; ++ct) {
        const int col = ct * 16 + ln;           // this lane's column this tile
        const int jl  = lab[col];

        // B fragment as loaded here (same k order as A): lane holds column (lane&15), 16 consecutive
        // K values starting at half*16 -> contiguous in row-major F.
        const size_t bbase = (size_t)col * K_DIM + half * 16;
        v16bf bh[4], bl[4];
#pragma unroll
        for (int c = 0; c < 4; ++c) {
            bh[c] = *(const v16bf*)(hi + bbase + c * 32);
            bl[c] = *(const v16bf*)(lo + bbase + c * 32);
        }

        // emulated-fp32 GEMM: hi*hi + hi*lo + lo*hi, f32 accumulate.
        // Result is already the base-2 logit s2 = (sim/T)*log2(e).
        v8f acc = {};
#pragma unroll
        for (int c = 0; c < 4; ++c) {
            acc = __builtin_amdgcn_wmma_f32_16x16x32_bf16(
                      false, ah[c].v, false, bh[c], (short)0, acc, false, false);
            acc = __builtin_amdgcn_wmma_f32_16x16x32_bf16(
                      false, ah[c].v, false, bl[c], (short)0, acc, false, false);
            acc = __builtin_amdgcn_wmma_f32_16x16x32_bf16(
                      false, al[c].v, false, bh[c], (short)0, acc, false, false);
        }

        // Online softmax update, single exp2 per element, select-based
        // (no fmin/fmax canonicalization). Max over ALL entries (incl.
        // diagonal); sum only where labels differ (diagonal auto-excluded).
#pragma unroll
        for (int r = 0; r < 8; ++r) {
            const float s   = acc[r];
            const float mo  = mrun[r];
            const float d   = mo - s;
            const bool  up  = d < 0.0f;                   // s is new max
            const float f   = fast_exp2(-__builtin_fabsf(d)); // free |.| modifier
            const float scl = up ? f : 1.0f;              // rescale old sum
            const float t   = up ? 1.0f : f;              // new term value
            const float add = (jl != rlab[r]) ? t : 0.0f; // neg-label mask
            srun[r] = srun[r] * scl + add;
            mrun[r] = up ? s : mo;
        }
    }

    // ---- combine the 16 lanes that share each row; store partials ---------
#pragma unroll
    for (int r = 0; r < 8; ++r) {
        float m = mrun[r], s = srun[r];
#pragma unroll
        for (int off = 1; off < 16; off <<= 1) {
            const float mo = __shfl_xor(m, off, 16);
            const float so = __shfl_xor(s, off, 16);
            const float d  = m - mo;
            const bool  up = d < 0.0f;                    // mo is larger
            const float f  = fast_exp2(-__builtin_fabsf(d));
            // larger side keeps scale 1, smaller side scaled by f
            s = up ? (s * f + so) : (so * f + s);
            m = up ? mo : m;
        }
        if (ln == 0) {
            const int row = rowBase + half * 8 + r;
            pm[row * CCHUNKS + chunk] = m;
            ps[row * CCHUNKS + chunk] = s;
        }
    }
}

// ---------------------------------------------------------------------------
// Merge the CCHUNKS partials per row, compute per-row loss, reduce to the
// final scalar. Deterministic (fixed partition + fixed-order tree).
// ---------------------------------------------------------------------------
__global__ __launch_bounds__(1024)
void k_reduce(const int*   __restrict__ lab, const float* __restrict__ iou,
              const int*   __restrict__ cnt,
              const float* __restrict__ pm,  const float* __restrict__ ps,
              float* __restrict__ out)
{
    __shared__ float sl[1024];
    __shared__ float sk[1024];
    float a = 0.0f, b = 0.0f;
    for (int i = threadIdx.x; i < M_ROWS; i += 1024) {
        float m[CCHUNKS];
        float M = pm[i * CCHUNKS + 0];
        m[0] = M;
#pragma unroll
        for (int c = 1; c < CCHUNKS; ++c) {
            m[c] = pm[i * CCHUNKS + c];
            M = fmaxf(M, m[c]);
        }
        float S = 0.0f;
#pragma unroll
        for (int c = 0; c < CCHUNKS; ++c)
            S += ps[i * CCHUNKS + c] * fast_exp2(m[c] - M);

        const float pc   = (float)(cnt[lab[i]] - 1);
        const float loss = LAMDA_W * S / pc;      // = -log_prob
        if (iou[i] >= IOU_THR) { a += loss; b += 1.0f; }
    }
    sl[threadIdx.x] = a; sk[threadIdx.x] = b;
    __syncthreads();
    for (int s = 512; s > 0; s >>= 1) {
        if (threadIdx.x < s) {
            sl[threadIdx.x] += sl[threadIdx.x + s];
            sk[threadIdx.x] += sk[threadIdx.x + s];
        }
        __syncthreads();
    }
    if (threadIdx.x == 0) out[0] = sl[0] / sk[0];
}

// ---------------------------------------------------------------------------
extern "C" void kernel_launch(void* const* d_in, const int* in_sizes, int n_in,
                              void* d_out, int out_size, void* d_ws, size_t ws_size,
                              hipStream_t stream)
{
    const float* feat = (const float*)d_in[0];
    const int*   lab  = (const int*)  d_in[1];
    const float* iou  = (const float*)d_in[2];

    char*   base = (char*)d_ws;
    __bf16* hi   = (__bf16*)base;
    __bf16* lo   = hi + (size_t)M_ROWS * K_DIM;
    int*    cnt  = (int*)  (base + WS_CNT_OFF);
    float*  pm   = (float*)(base + WS_PM_OFF);
    float*  ps   = (float*)(base + WS_PS_OFF);
    float*  out  = (float*)d_out;

    k_init  <<<1, 32, 0, stream>>>(cnt);
    k_hist  <<<(M_ROWS + 255) / 256, 256, 0, stream>>>(lab, cnt);
    k_split <<<(M_ROWS * K_DIM + 255) / 256, 256, 0, stream>>>(feat, hi, lo);
    k_main  <<<(M_ROWS / 128) * CCHUNKS, 256, 0, stream>>>(hi, lo, lab, pm, ps);
    k_reduce<<<1, 1024, 0, stream>>>(lab, iou, cnt, pm, ps, out);
}
